// Head_74500502716418
// MI455X (gfx1250) — compile-verified, matrix operand order checked by reading, not run on hardware
//
#include <hip/hip_runtime.h>
#include <hip/hip_bf16.h>

#define TSEQ  128
#define NEMBD 512
#define HEAD  64

typedef __attribute__((ext_vector_type(16))) __bf16       v16bf;
typedef __attribute__((ext_vector_type(8)))  float        v8f;
typedef __attribute__((ext_vector_type(4)))  unsigned int v4u;

// LDS layout (bytes), total 155136 (<160KB):
//   qs:     0      .. 18432   (128 x 72  u16, bf16 row-major)
//   ks:     18432  .. 36864   (128 x 72  u16, bf16 row-major)
//   vT:     36864  .. 54272   (64  x 136 u16, bf16 transposed: vT[h][t])
//   sc:     54272  .. 120320  (128 x 129 f32 scores)
//   wm:     120320 .. 155136  (128 x 136 u16 softmax weights)
// Phase-1 staging overlaps [0 .. 46080):
//   xs:  0     .. 18432  (128 x 72 u16)
//   wsT: 18432 .. 46080  (3 x 64 x 72 u16, W chunk transposed)

__device__ __forceinline__ unsigned short f32_to_bf16(float f) {
  unsigned int u = __float_as_uint(f);
  u += 0x7FFFu + ((u >> 16) & 1u);   // round-to-nearest-even
  return (unsigned short)(u >> 16);
}

__device__ __forceinline__ unsigned int pack_bf16x2(float lo, float hi) {
  return (unsigned int)f32_to_bf16(lo) | ((unsigned int)f32_to_bf16(hi) << 16);
}

// Load a 16x32 bf16 A-style fragment (per ISA 16-bit A layout):
// lane m (0-15): K = kbase+{0..7,16..23}; lane m+16: +8. Two ds_load_b128.
__device__ __forceinline__ v16bf load_frag(const unsigned short* base, int row,
                                           int stride, int kbase, int lane) {
  const int m = lane & 15, hl = lane >> 4;
  const unsigned short* p = base + (row + m) * stride + kbase + hl * 8;
  union { v16bf bf; v4u u[2]; } f;
  f.u[0] = *(const v4u*)(p);
  f.u[1] = *(const v4u*)(p + 16);
  return f.bf;
}

__global__ __launch_bounds__(256) void attn_head_fused(
    const float* __restrict__ x,  const float* __restrict__ Wq,
    const float* __restrict__ Wk, const float* __restrict__ Wv,
    float* __restrict__ out) {
  __shared__ alignas(16) unsigned char lds[155136];
  unsigned short* qs  = (unsigned short*)(lds + 0);
  unsigned short* ks  = (unsigned short*)(lds + 18432);
  unsigned short* vT  = (unsigned short*)(lds + 36864);
  float*          sc  = (float*)         (lds + 54272);
  unsigned short* wm  = (unsigned short*)(lds + 120320);
  unsigned short* xs  = (unsigned short*)(lds + 0);       // phase-1 staging
  unsigned short* wsT = (unsigned short*)(lds + 18432);   // phase-1 staging

  const int b    = blockIdx.x;
  const int tid  = threadIdx.x;
  const int lane = tid & 31;
  const int wv   = tid >> 5;     // wave id 0..7 -> owns row-tile wv
  const int hl   = lane >> 4;
  const int nn   = lane & 15;

  const float* xb = x + (size_t)b * TSEQ * NEMBD;
  const float* W[3] = {Wq, Wk, Wv};

  v8f acc[3][4] = {};            // q/k/v accumulators: 3 mats x 4 col-tiles

  // ---------------- Phase 1: q,k,v = x @ W* ----------------
  for (int kc = 0; kc < NEMBD / 64; ++kc) {
    __syncthreads();
    // stage x[:, kc*64 .. +64) as bf16 pairs (b64 global read, b32 LDS write)
    #pragma unroll
    for (int i = 0; i < 16; ++i) {
      int idx = tid + i * 256;                 // 0..4095 (pairs)
      int row = idx >> 5, col = (idx & 31) * 2;
      float2 v = *(const float2*)(xb + row * NEMBD + kc * 64 + col);
      *(unsigned int*)(xs + row * 72 + col) = pack_bf16x2(v.x, v.y);
    }
    // stage W chunks transposed: wsT[m][n*72 + k] = W[m][(kc*64+k)*HEAD + n]
    // pack two k's per b32 LDS write
    #pragma unroll
    for (int m = 0; m < 3; ++m) {
      #pragma unroll
      for (int i = 0; i < 8; ++i) {
        int idx = tid + i * 256;               // 0..2047 (pairs along k)
        int cn = idx & 63, kk = (idx >> 6) * 2;
        float w0 = W[m][(kc * 64 + kk)     * HEAD + cn];
        float w1 = W[m][(kc * 64 + kk + 1) * HEAD + cn];
        *(unsigned int*)(wsT + m * 64 * 72 + cn * 72 + kk) = pack_bf16x2(w0, w1);
      }
    }
    __syncthreads();
    // software-pipelined fragment feed: 24 WMMAs, B frag i+1 issued before WMMA i
    v16bf a[2];
    a[0] = load_frag(xs, wv * 16, 72, 0,  lane);
    a[1] = load_frag(xs, wv * 16, 72, 32, lane);
    v16bf bcur = load_frag(wsT, 0, 72, 0, lane);  // (sub=0, m=0, c=0)
    #pragma unroll
    for (int it = 0; it < 24; ++it) {
      const int sub = it / 12, rem = it % 12, m = rem / 4, c = rem % 4;
      v16bf bnext = bcur;
      if (it < 23) {
        const int jt = it + 1;
        const int sub2 = jt / 12, rem2 = jt % 12, m2 = rem2 / 4, c2 = rem2 % 4;
        bnext = load_frag(wsT + m2 * 64 * 72, c2 * 16, 72, sub2 * 32, lane);
      }
      acc[m][c] = __builtin_amdgcn_wmma_f32_16x16x32_bf16(
          false, a[sub], false, bcur, (short)0, acc[m][c], false, false);
      bcur = bnext;
    }
  }
  __syncthreads();

  // ---------------- Phase 2: spill q,k row-major; v transposed ----------------
  #pragma unroll
  for (int c = 0; c < 4; ++c) {
    #pragma unroll
    for (int r = 0; r < 8; ++r) {
      int t = wv * 16 + r + hl * 8;
      qs[t * 72 + c * 16 + nn]    = f32_to_bf16(acc[0][c][r]);
      ks[t * 72 + c * 16 + nn]    = f32_to_bf16(acc[1][c][r]);
      vT[(c * 16 + nn) * 136 + t] = f32_to_bf16(acc[2][c][r]);
    }
  }
  __syncthreads();

  // ---------------- Phase 3: scores = q @ k^T ----------------
  v8f sacc[8] = {};
  {
    v16bf a[2];
    a[0] = load_frag(qs, wv * 16, 72, 0,  lane);
    a[1] = load_frag(qs, wv * 16, 72, 32, lane);
    v16bf bcur = load_frag(ks, 0, 72, 0, lane);
    #pragma unroll
    for (int it = 0; it < 16; ++it) {
      const int sub = it / 8, c = it % 8;
      v16bf bnext = bcur;
      if (it < 15) {
        const int jt = it + 1;
        bnext = load_frag(ks, (jt % 8) * 16, 72, (jt / 8) * 32, lane);
      }
      sacc[c] = __builtin_amdgcn_wmma_f32_16x16x32_bf16(
          false, a[sub], false, bcur, (short)0, sacc[c], false, false);
      bcur = bnext;
    }
  }
  #pragma unroll
  for (int c = 0; c < 8; ++c) {
    #pragma unroll
    for (int r = 0; r < 8; ++r)
      sc[(wv * 16 + r + hl * 8) * 129 + c * 16 + nn] = sacc[c][r];
  }
  __syncthreads();

  // ---------------- Phase 4: causal softmax (row per thread) ----------------
  if (tid < TSEQ) {
    const int r = tid;
    float* srow = sc + r * 129;
    float mx = -3.0e38f;
    for (int s = 0; s <= r; ++s) {
      float v = srow[s] * 0.125f;       // HEAD^-0.5
      srow[s] = v;
      mx = fmaxf(mx, v);
    }
    float sum = 0.0f;
    for (int s = 0; s <= r; ++s) {
      float e = __expf(srow[s] - mx);
      srow[s] = e;
      sum += e;
    }
    float inv = 1.0f / sum;
    unsigned short* wrow = wm + r * 136;
    for (int s = 0; s <= r; ++s)  wrow[s] = f32_to_bf16(srow[s] * inv);
    for (int s = r + 1; s < TSEQ; ++s) wrow[s] = 0;
  }
  __syncthreads();

  // ---------------- Phase 5: out = w @ v ----------------
  v8f oacc[4] = {};
  {
    v16bf a[4];
    #pragma unroll
    for (int sub = 0; sub < 4; ++sub)
      a[sub] = load_frag(wm, wv * 16, 136, sub * 32, lane);
    v16bf bcur = load_frag(vT, 0, 136, 0, lane);
    #pragma unroll
    for (int it = 0; it < 16; ++it) {
      const int sub = it / 4, c = it % 4;
      v16bf bnext = bcur;
      if (it < 15) {
        const int jt = it + 1;
        bnext = load_frag(vT, (jt % 4) * 16, 136, (jt / 4) * 32, lane);
      }
      oacc[c] = __builtin_amdgcn_wmma_f32_16x16x32_bf16(
          false, a[sub], false, bcur, (short)0, oacc[c], false, false);
      bcur = bnext;
    }
  }
  float* ob = out + (size_t)b * TSEQ * HEAD;
  #pragma unroll
  for (int c = 0; c < 4; ++c) {
    #pragma unroll
    for (int r = 0; r < 8; ++r) {
      int t = wv * 16 + r + hl * 8;
      ob[t * HEAD + c * 16 + nn] = oacc[c][r];
    }
  }
}

extern "C" void kernel_launch(void* const* d_in, const int* in_sizes, int n_in,
                              void* d_out, int out_size, void* d_ws, size_t ws_size,
                              hipStream_t stream) {
  (void)in_sizes; (void)n_in; (void)out_size; (void)d_ws; (void)ws_size;
  const float* x  = (const float*)d_in[0];
  const float* Wq = (const float*)d_in[1];
  const float* Wk = (const float*)d_in[2];
  const float* Wv = (const float*)d_in[3];
  float* out = (float*)d_out;
  const int B = 1024;
  attn_head_fused<<<dim3(B), dim3(256), 0, stream>>>(x, Wq, Wk, Wv, out);
}
